// Head_79482664780310
// MI455X (gfx1250) — matrix operand order repaired and checked by reading, not compiled, not run on hardware
//
#include <hip/hip_runtime.h>
#include <hip/hip_bf16.h>

typedef __attribute__((ext_vector_type(16))) __bf16          v16bf;
typedef __attribute__((ext_vector_type(8)))  float           v8f;
typedef __attribute__((ext_vector_type(8)))  unsigned short  u16x8;
typedef __attribute__((ext_vector_type(16))) unsigned short  v16u;
typedef __attribute__((ext_vector_type(4)))  unsigned int    tdm_v4u;
typedef __attribute__((ext_vector_type(8)))  int             tdm_v8i;
typedef __attribute__((ext_vector_type(4)))  int             tdm_v4i;

#define BB 4
#define TT 2048
#define CC 1024
#define HSZ 64

__device__ __forceinline__ unsigned short f2bf(float f) {
    unsigned int u = __float_as_uint(f);
    u += 0x7FFFu + ((u >> 16) & 1u);          // round-to-nearest-even
    return (unsigned short)(u >> 16);
}

union FragU { v16u u; u16x8 h[2]; v16bf bf; unsigned short s[16]; };

// ---------------------------------------------------------------------------
// TDM: async 2D bf16 tile (tile1 rows x tile0 elems) -> LDS, in-order,
// tracked by TENSORcnt. dims are "remaining" extents so OOB reads return 0.
// D# packing per CDNA5 ISA/S8.4 (2D: groups 2,3 = 0).
// 6-arg builtin form (amdgpu-toolchain clang-23).
// ---------------------------------------------------------------------------
__device__ __forceinline__ void tdm_load_2d(unsigned lds_off, const void* gaddr,
                                            unsigned dim0, unsigned dim1,
                                            unsigned tile0, unsigned tile1,
                                            unsigned stride0) {
    unsigned long long ga = (unsigned long long)(uintptr_t)gaddr;
    tdm_v4u g0;
    g0[0] = 1u;                                              // count=1, user D#
    g0[1] = lds_off;                                         // lds_addr
    g0[2] = (unsigned)(ga & 0xFFFFFFFFu);                    // global_addr lo
    g0[3] = (unsigned)((ga >> 32) & 0x01FFFFFFu) | 0x80000000u; // hi | type=2
    tdm_v8i g1;
    g1[0] = 0x00010000;                                      // data_size=1 (2B)
    g1[1] = (int)((dim0 & 0xFFFFu) << 16);                   // tensor_dim0 lo
    g1[2] = (int)((dim0 >> 16) | ((dim1 & 0xFFFFu) << 16));  // dim0 hi|dim1 lo
    g1[3] = (int)((dim1 >> 16) | (tile0 << 16));             // dim1 hi|tile0
    g1[4] = (int)(tile1 & 0xFFFFu);                          // tile1, tile2=0
    g1[5] = (int)stride0;                                    // dim0_stride lo32
    g1[6] = 0;                                               // stride hi, dim1_stride
    g1[7] = 0;
    tdm_v4i z4 = {0, 0, 0, 0};
    tdm_v8i z8 = {0, 0, 0, 0, 0, 0, 0, 0};
    __builtin_amdgcn_tensor_load_to_lds(g0, g1, z4, z4, z8, 0);
}

__device__ __forceinline__ unsigned lds_offset(const void* p) {
    return (unsigned)(uintptr_t)p;   // LDS aperture: addr[31:0] = LDS offset
}

// ---------------------------------------------------------------------------
// Kernel 0: transpose + fp32->bf16 convert weights: Wt[w][h][k]  (w=q,k,v)
// ---------------------------------------------------------------------------
__global__ void wt_kernel(const float* __restrict__ Wq,
                          const float* __restrict__ Wk,
                          const float* __restrict__ Wv,
                          unsigned short* __restrict__ Wt) {
    int idx = blockIdx.x * blockDim.x + threadIdx.x;   // over 64*1024
    int w   = blockIdx.y;
    const float* W = (w == 0) ? Wq : (w == 1) ? Wk : Wv;
    int h = idx >> 10;
    int k = idx & 1023;
    Wt[((size_t)w * HSZ + h) * CC + k] = f2bf(W[(size_t)k * HSZ + h]);
}

// ---------------------------------------------------------------------------
// Kernel 1: fused QKV projection. One wave per 16-row tile of B*T.
// x read once; Q,K row-major bf16 [B*T][64]; V transposed bf16 [B][64][T].
// All 12 B fragments live in distinct VGPRs so the 24 b128 loads issue as one
// clause and the WMMA drain overlaps memory latency.
// ---------------------------------------------------------------------------
__global__ void __launch_bounds__(32)
qkv_kernel(const float* __restrict__ x, const unsigned short* __restrict__ Wt,
           unsigned short* __restrict__ Qb, unsigned short* __restrict__ Kb,
           unsigned short* __restrict__ Vt) {
    const int lane = threadIdx.x;
    const int mt   = blockIdx.x;            // 0..(B*T/16 - 1)
    const int half = lane >> 4;
    const int akb  = half * 8;              // A-fragment K base
    const int bkb  = half * 8;              // B-fragment K base
    const int ncol = lane & 15;
    const int arow = mt * 16 + ncol;

    v8f acc[3][4] = {};

    for (int k0 = 0; k0 < CC; k0 += 32) {
        const float* xp = x + (size_t)arow * CC + k0 + akb;
        if (k0 + 64 < CC) __builtin_prefetch(xp + 64, 0, 0);   // next x chunk

        // ---- A fragment from x (fp32 -> bf16), two contiguous 8-float runs
        FragU a;
#pragma unroll
        for (int j = 0; j < 8; ++j) {
            a.s[j]     = f2bf(xp[j]);
            a.s[8 + j] = f2bf(xp[16 + j]);
        }
        // ---- all 12 B fragments up-front (distinct regs -> one load clause)
        FragU bf[3][4];
#pragma unroll
        for (int w = 0; w < 3; ++w)
#pragma unroll
            for (int t = 0; t < 4; ++t) {
                const unsigned short* wp =
                    Wt + ((size_t)w * HSZ + t * 16 + ncol) * CC + k0 + bkb;
                bf[w][t].h[0] = *(const u16x8*)(wp);
                bf[w][t].h[1] = *(const u16x8*)(wp + 16);
            }
        // ---- drain 12 WMMAs
#pragma unroll
        for (int w = 0; w < 3; ++w)
#pragma unroll
            for (int t = 0; t < 4; ++t)
                acc[w][t] = __builtin_amdgcn_wmma_f32_16x16x32_bf16(
                    false, a.bf, false, bf[w][t].bf, (short)0, acc[w][t],
                    false, false);
    }

    // ---- store: C-layout element (vgpr r, lane) = (M = half*8+r, N = ncol)
#pragma unroll
    for (int t = 0; t < 4; ++t)
#pragma unroll
        for (int r = 0; r < 8; ++r) {
            int m = half * 8 + r;
            size_t g = (size_t)mt * 16 + m;
            int col = t * 16 + ncol;
            Qb[g * HSZ + col] = f2bf(acc[0][t][r]);
            Kb[g * HSZ + col] = f2bf(acc[1][t][r]);
            size_t bb = g / TT, tpos = g % TT;
            Vt[(bb * HSZ + col) * TT + tpos] = f2bf(acc[2][t][r]);
        }
}

// ---------------------------------------------------------------------------
// Kernel 2: flash attention, one wave per (batch, 16-query tile).
// K/V blocks stream through double-buffered LDS via TDM (async, TENSORcnt),
// prefetching block i+1 while computing block i.
// ---------------------------------------------------------------------------
__global__ void __launch_bounds__(32)
attn_kernel(const unsigned short* __restrict__ Qb,
            const unsigned short* __restrict__ Kb,
            const unsigned short* __restrict__ Vt,
            float* __restrict__ out) {
    __shared__ __align__(16) unsigned short kbuf[2][32 * HSZ];  // keys x hs
    __shared__ __align__(16) unsigned short vbuf[2][HSZ * 32];  // hs x keys
    __shared__ __align__(16) unsigned short pbuf[16 * 40];      // padded P

    const int lane = threadIdx.x;
    const int qt   = blockIdx.x;            // 0..T/16-1
    const int b    = blockIdx.y;            // 0..B-1
    const int half = lane >> 4;
    const int akb  = half * 8;
    const int bkb  = half * 8;
    const int ncol = lane & 15;
    const size_t bT = (size_t)b * TT;
    const float scale = 0.125f;             // 1/sqrt(64)
    const int nblk = qt / 2 + 1;            // 32-key blocks (causal)

    // ---- issue TDM for block 0, then load Q while it flies
    tdm_load_2d(lds_offset(&kbuf[0][0]), Kb + (bT + 0) * HSZ,
                HSZ, TT, HSZ, 32, HSZ);
    tdm_load_2d(lds_offset(&vbuf[0][0]), Vt + (size_t)b * HSZ * TT + 0,
                TT, HSZ, 32, HSZ, TT);

    v16bf qa[2];
    {
        const unsigned short* qp = Qb + (bT + qt * 16 + ncol) * HSZ;
#pragma unroll
        for (int c = 0; c < 2; ++c) {
            FragU f;
            f.h[0] = *(const u16x8*)(qp + c * 32 + akb);
            f.h[1] = *(const u16x8*)(qp + c * 32 + akb + 16);
            qa[c] = f.bf;
        }
    }

    float m_i[8], l_i[8];
    v8f o[4] = {};
#pragma unroll
    for (int r = 0; r < 8; ++r) { m_i[r] = -1e30f; l_i[r] = 0.0f; }

    for (int i = 0; i < nblk; ++i) {
        const int kb0 = i * 32;
        const int buf = i & 1;

        __builtin_amdgcn_s_wait_tensorcnt(0);      // block i resident in LDS
        if (i + 1 < nblk) {                        // prefetch block i+1
            int kn = kb0 + 32;
            tdm_load_2d(lds_offset(&kbuf[buf ^ 1][0]), Kb + (bT + kn) * HSZ,
                        HSZ, TT - kn, HSZ, 32, HSZ);
            tdm_load_2d(lds_offset(&vbuf[buf ^ 1][0]),
                        Vt + (size_t)b * HSZ * TT + kn,
                        TT - kn, HSZ, 32, HSZ, TT);
        }

        // ---- S = Q K^T for 32 keys (two 16-key subtiles), K from LDS
        v8f s[2];
#pragma unroll
        for (int sub = 0; sub < 2; ++sub) {
            const unsigned short* kp = &kbuf[buf][(sub * 16 + ncol) * HSZ];
            v8f z = {};
#pragma unroll
            for (int c = 0; c < 2; ++c) {
                FragU f;
                f.h[0] = *(const u16x8*)(kp + c * 32 + bkb);
                f.h[1] = *(const u16x8*)(kp + c * 32 + bkb + 16);
                z = __builtin_amdgcn_wmma_f32_16x16x32_bf16(
                    false, qa[c], false, f.bf, (short)0, z, false, false);
            }
            s[sub] = z;
        }

        // ---- causal mask + online softmax (rows span 16-lane halves)
#pragma unroll
        for (int r = 0; r < 8; ++r) {
            int qg = qt * 16 + half * 8 + r;
            float s0 = s[0][r] * scale;
            float s1 = s[1][r] * scale;
            if (kb0 + ncol      > qg) s0 = -1e30f;
            if (kb0 + 16 + ncol > qg) s1 = -1e30f;
            float mx = fmaxf(s0, s1);
#pragma unroll
            for (int d = 1; d < 16; d <<= 1) mx = fmaxf(mx, __shfl_xor(mx, d));
            float mnew = fmaxf(m_i[r], mx);
            float p0 = __expf(s0 - mnew);
            float p1 = __expf(s1 - mnew);
            float rs = p0 + p1;
#pragma unroll
            for (int d = 1; d < 16; d <<= 1) rs += __shfl_xor(rs, d);
            float alpha = __expf(m_i[r] - mnew);
            l_i[r] = l_i[r] * alpha + rs;
            m_i[r] = mnew;
#pragma unroll
            for (int t = 0; t < 4; ++t) o[t][r] *= alpha;
            int m = half * 8 + r;
            pbuf[m * 40 + ncol]      = f2bf(p0);
            pbuf[m * 40 + 16 + ncol] = f2bf(p1);
        }
        __syncthreads();   // single-wave WG: compiler fence, hw NOP

        // ---- reload P as A-fragment (16x32, K=keys)
        FragU pf;
        {
            const unsigned short* pp = pbuf + ncol * 40 + akb;
            pf.h[0] = *(const u16x8*)(pp);
            pf.h[1] = *(const u16x8*)(pp + 16);
        }

        // ---- O += P @ V, V from LDS (contiguous along keys)
#pragma unroll
        for (int t = 0; t < 4; ++t) {
            const unsigned short* vp = &vbuf[buf][(t * 16 + ncol) * 32 + bkb];
            FragU vf;
            vf.h[0] = *(const u16x8*)(vp);
            vf.h[1] = *(const u16x8*)(vp + 16);
            o[t] = __builtin_amdgcn_wmma_f32_16x16x32_bf16(
                false, pf.bf, false, vf.bf, (short)0, o[t], false, false);
        }
        __syncthreads();
    }

    // ---- normalize + store fp32 output
#pragma unroll
    for (int t = 0; t < 4; ++t)
#pragma unroll
        for (int r = 0; r < 8; ++r) {
            int m = half * 8 + r;
            out[(bT + qt * 16 + m) * HSZ + t * 16 + ncol] = o[t][r] / l_i[r];
        }
}

// ---------------------------------------------------------------------------
extern "C" void kernel_launch(void* const* d_in, const int* in_sizes, int n_in,
                              void* d_out, int out_size, void* d_ws, size_t ws_size,
                              hipStream_t stream) {
    (void)in_sizes; (void)n_in; (void)out_size; (void)ws_size;
    const float* x  = (const float*)d_in[0];
    const float* Wq = (const float*)d_in[1];
    const float* Wk = (const float*)d_in[2];
    const float* Wv = (const float*)d_in[3];

    unsigned short* ws = (unsigned short*)d_ws;
    unsigned short* Wt = ws;                                   // 3*64*1024
    unsigned short* Qb = Wt + 3 * HSZ * CC;                    // B*T*64
    unsigned short* Kb = Qb + (size_t)BB * TT * HSZ;
    unsigned short* Vt = Kb + (size_t)BB * TT * HSZ;

    wt_kernel<<<dim3((HSZ * CC) / 256, 3), 256, 0, stream>>>(Wq, Wk, Wv, Wt);
    qkv_kernel<<<dim3((BB * TT) / 16), 32, 0, stream>>>(x, Wt, Qb, Kb, Vt);
    attn_kernel<<<dim3(TT / 16, BB), 32, 0, stream>>>(Qb, Kb, Vt, (float*)d_out);
}
